// MQA_20100446945550
// MI455X (gfx1250) — matrix operand order repaired and checked by reading, not compiled, not run on hardware
//
#include <hip/hip_runtime.h>
#include <hip/hip_bf16.h>

#define D_MODEL   2048
#define NUM_HEADS 16
#define HEAD_DIM  128
#define SEQ       2048
#define BATCH     2

typedef __attribute__((ext_vector_type(16))) __bf16 v16bf;
typedef __attribute__((ext_vector_type(8)))  float  v8f;

union FragAB {
    v16bf v;
    uint4 u[2];
};

__device__ __forceinline__ v8f wmma_bf16(const FragAB& a, const FragAB& b, v8f c) {
    return __builtin_amdgcn_wmma_f32_16x16x32_bf16(false, a.v, false, b.v,
                                                   (short)0, c, false, false);
}

__device__ __forceinline__ v8f v8f_zero() {
    v8f z;
#pragma unroll
    for (int i = 0; i < 8; ++i) z[i] = 0.0f;
    return z;
}

// ---- CDNA5 async copy engine: global -> LDS without VGPR round trip ----
// (tracked by ASYNCcnt; ds address = low 32 bits of the generic LDS pointer)
__device__ __forceinline__ void async_b128(unsigned int lds_byte, const __bf16* g) {
    asm volatile("global_load_async_to_lds_b128 %0, %1, off"
                 :: "v"(lds_byte), "v"((unsigned long long)(uintptr_t)g)
                 : "memory");
}
__device__ __forceinline__ void wait_async0() {
    asm volatile("s_wait_asynccnt 0x0" ::: "memory");
}

// ---------------------------------------------------------------------------
// Elementwise fp32 -> bf16 (vectorized x4)
// ---------------------------------------------------------------------------
__global__ __launch_bounds__(256)
void cvt_f32_to_bf16(const float* __restrict__ in, __bf16* __restrict__ out, int n4)
{
    const int i = blockIdx.x * 256 + threadIdx.x;
    if (i >= n4) return;
    float4 f = ((const float4*)in)[i];
    union { __bf16 h[4]; uint2 u; } r;
    r.h[0] = (__bf16)f.x; r.h[1] = (__bf16)f.y;
    r.h[2] = (__bf16)f.z; r.h[3] = (__bf16)f.w;
    ((uint2*)out)[i] = r.u;
}

// ---------------------------------------------------------------------------
// Tiled transpose + convert: in fp32 [K][N] -> out bf16 [N][K]
// ---------------------------------------------------------------------------
__global__ __launch_bounds__(256)
void transpose_cvt_bf16(const float* __restrict__ in, __bf16* __restrict__ out,
                        int K, int N)
{
    __shared__ __bf16 t[32][33];
    const int tx = threadIdx.x, ty = threadIdx.y;
    const int k0 = blockIdx.y * 32, n0 = blockIdx.x * 32;
#pragma unroll
    for (int j = 0; j < 4; ++j)
        t[ty + j * 8][tx] = (__bf16)in[(size_t)(k0 + ty + j * 8) * N + n0 + tx];
    __syncthreads();
#pragma unroll
    for (int j = 0; j < 4; ++j)
        out[(size_t)(n0 + ty + j * 8) * K + k0 + tx] = t[tx][ty + j * 8];
}

// ---------------------------------------------------------------------------
// bf16 GEMM: C[M,N] = A[M,K] * BT[N,K]^T + bias[N]
// A row-major bf16, BT row-major bf16 (pre-transposed weight).
// Block tile 128x128, K-step 32, 8 waves (4 M x 2 N), wave tile 32x64.
// Double-buffered LDS fed by async-copy engine.
// TRANS_OUT: write C as [b][n][s] (per-batch transposed, for V^T).
// ---------------------------------------------------------------------------
template <typename TC, bool TRANS_OUT>
__global__ __launch_bounds__(256)
void gemm_bf16_wmma(const __bf16* __restrict__ A, const __bf16* __restrict__ BT,
                    const float* __restrict__ bias, TC* __restrict__ C,
                    int M, int N, int K)
{
    __shared__ __bf16 As[2][128 * 40];   // [row][k], padded stride 40
    __shared__ __bf16 Bs[2][128 * 40];   // [n][k],   padded stride 40

    const int tid  = threadIdx.x;
    const int lane = tid & 31;
    const int lo16 = lane & 15;
    const int hi   = lane >> 4;
    const int wave = tid >> 5;
    const int wm   = wave & 3;
    const int wn   = wave >> 2;

    const int m0 = blockIdx.y * 128;
    const int n0 = blockIdx.x * 128;

    const int row = tid >> 1, seg = tid & 1;   // staging assignment (32B each)

    auto stage = [&](int kt, int buf) {
        const __bf16* as = A + (size_t)(m0 + row) * K + kt * 32 + seg * 16;
        unsigned int ad = (unsigned int)(uintptr_t)&As[buf][row * 40 + seg * 16];
        async_b128(ad, as);
        async_b128(ad + 16, as + 8);
        const __bf16* bs = BT + (size_t)(n0 + row) * K + kt * 32 + seg * 16;
        unsigned int bd = (unsigned int)(uintptr_t)&Bs[buf][row * 40 + seg * 16];
        async_b128(bd, bs);
        async_b128(bd + 16, bs + 8);
    };

    v8f acc[2][4];
#pragma unroll
    for (int mi = 0; mi < 2; ++mi)
#pragma unroll
        for (int ni = 0; ni < 4; ++ni) acc[mi][ni] = v8f_zero();

    const int T = K >> 5;
    stage(0, 0);
    for (int i = 0; i < T; ++i) {
        const int cur = i & 1;
        wait_async0();          // tile i landed in LDS
        __syncthreads();        // visible to all waves; prev compute finished
        if (i + 1 < T) stage(i + 1, cur ^ 1);   // prefetch overlaps WMMAs

        FragAB a[2], b[4];
#pragma unroll
        for (int mi = 0; mi < 2; ++mi) {
            const __bf16* p = &As[cur][(wm * 32 + mi * 16 + lo16) * 40];
            const int base = hi * 8;
            a[mi].u[0] = *(const uint4*)(p + base);
            a[mi].u[1] = *(const uint4*)(p + 16 + base);
        }
#pragma unroll
        for (int ni = 0; ni < 4; ++ni) {
            const __bf16* p = &Bs[cur][(wn * 64 + ni * 16 + lo16) * 40 + hi * 8];
            b[ni].u[0] = *(const uint4*)(p);
            b[ni].u[1] = *(const uint4*)(p + 16);
        }
#pragma unroll
        for (int mi = 0; mi < 2; ++mi)
#pragma unroll
            for (int ni = 0; ni < 4; ++ni)
                acc[mi][ni] = wmma_bf16(a[mi], b[ni], acc[mi][ni]);
    }

    // ---- epilogue ----
#pragma unroll
    for (int mi = 0; mi < 2; ++mi) {
#pragma unroll
        for (int ni = 0; ni < 4; ++ni) {
            const int gn = n0 + wn * 64 + ni * 16 + lo16;
            const float bv = bias[gn];
#pragma unroll
            for (int r = 0; r < 8; ++r) {
                const int gm = m0 + wm * 32 + mi * 16 + r + hi * 8;
                const float val = acc[mi][ni][r] + bv;
                if constexpr (TRANS_OUT) {
                    // [b][n][s]: b = gm/SEQ, s = gm%SEQ
                    C[((size_t)(gm >> 11) * N + gn) * SEQ + (gm & (SEQ - 1))] = (TC)val;
                } else {
                    C[(size_t)gm * N + gn] = (TC)val;
                }
            }
        }
    }
}

// ---------------------------------------------------------------------------
// Flash MQA attention. One block = 128 query rows of one (b,h); 8 waves,
// 16 q-rows per wave. K tile [kv][d] and V^T tile [d][kv] staged via async
// copy engine (both straight copies: vhT is produced pre-transposed).
// ---------------------------------------------------------------------------
__global__ __launch_bounds__(256)
void mqa_attention_wmma(const __bf16* __restrict__ qh,
                        const __bf16* __restrict__ kh,
                        const __bf16* __restrict__ vhT,
                        __bf16* __restrict__ ctx)
{
    extern __shared__ __bf16 smem[];
    __bf16* Kl = smem;                    // [128][136]
    __bf16* Vt = smem + 128 * 136;        // [128][136]
    __bf16* Pl = smem + 2 * 128 * 136;    // [8][16][136]

    const int tid  = threadIdx.x;
    const int lane = tid & 31;
    const int lo16 = lane & 15;
    const int hi   = lane >> 4;
    const int wave = tid >> 5;

    const int q0 = blockIdx.x * 128;
    const int h  = blockIdx.y;
    const int b  = blockIdx.z;

    const float scale = 0.08838834764831845f;    // 1/sqrt(128)
    const float L2E   = 1.4426950408889634f;

    // ---- Q fragments, loaded once ----
    FragAB qa[4];
    {
        const int qrow = q0 + wave * 16 + lo16;
        const __bf16* qp = qh + (size_t)(b * SEQ + qrow) * D_MODEL + h * HEAD_DIM;
#pragma unroll
        for (int ks = 0; ks < 4; ++ks) {
            const int base = ks * 32 + hi * 8;
            qa[ks].u[0] = *(const uint4*)(qp + base);
            qa[ks].u[1] = *(const uint4*)(qp + base + 16);
        }
    }

    float mrow[8], lrow[8];
    v8f   oacc[8];
#pragma unroll
    for (int r = 0; r < 8; ++r) { mrow[r] = -1.0e30f; lrow[r] = 0.0f; oacc[r] = v8f_zero(); }

    __bf16* Pw = Pl + wave * 16 * 136;
    const int srow = tid >> 1, shalf = tid & 1;   // staging assignment

    for (int kv0 = 0; kv0 < SEQ; kv0 += 128) {
        __syncthreads();   // everyone done reading previous K/V tiles
        {   // K tile: Kl[kv][d]  <- kh[(b,kv0+kv)][d]   (128B per thread)
            const __bf16* src = kh + (size_t)(b * SEQ + kv0 + srow) * HEAD_DIM + shalf * 64;
            unsigned int dst = (unsigned int)(uintptr_t)&Kl[srow * 136 + shalf * 64];
#pragma unroll
            for (int i = 0; i < 8; ++i) async_b128(dst + i * 16, src + i * 8);
        }
        {   // V^T tile: Vt[d][kv] <- vhT[b][d][kv0+kv]  (128B per thread)
            const __bf16* src = vhT + ((size_t)b * HEAD_DIM + srow) * SEQ + kv0 + shalf * 64;
            unsigned int dst = (unsigned int)(uintptr_t)&Vt[srow * 136 + shalf * 64];
#pragma unroll
            for (int i = 0; i < 8; ++i) async_b128(dst + i * 16, src + i * 8);
        }
        wait_async0();
        __syncthreads();

        // ---- S = Q * K^T ----
        v8f sc[8];
#pragma unroll
        for (int f = 0; f < 8; ++f) sc[f] = v8f_zero();
#pragma unroll
        for (int f = 0; f < 8; ++f) {
#pragma unroll
            for (int ks = 0; ks < 4; ++ks) {
                FragAB kb;
                const __bf16* p = Kl + (f * 16 + lo16) * 136 + ks * 32 + hi * 8;
                kb.u[0] = *(const uint4*)(p);
                kb.u[1] = *(const uint4*)(p + 16);
                sc[f] = wmma_bf16(qa[ks], kb, sc[f]);
            }
        }
#pragma unroll
        for (int f = 0; f < 8; ++f)
#pragma unroll
            for (int r = 0; r < 8; ++r) sc[f][r] *= scale;

        // ---- online softmax (row = r + 8*hi across 16 lanes) ----
#pragma unroll
        for (int r = 0; r < 8; ++r) {
            float tm = -1.0e30f;
#pragma unroll
            for (int f = 0; f < 8; ++f) tm = fmaxf(tm, sc[f][r]);
#pragma unroll
            for (int off = 1; off < 16; off <<= 1)
                tm = fmaxf(tm, __shfl_xor(tm, off, 32));
            const float mn    = fmaxf(mrow[r], tm);
            const float alpha = __builtin_amdgcn_exp2f((mrow[r] - mn) * L2E);
            mrow[r] = mn;
            float rs = 0.0f;
#pragma unroll
            for (int f = 0; f < 8; ++f) {
                const float p = __builtin_amdgcn_exp2f((sc[f][r] - mn) * L2E);
                sc[f][r] = p;
                rs += p;
            }
#pragma unroll
            for (int off = 1; off < 16; off <<= 1)
                rs += __shfl_xor(rs, off, 32);
            lrow[r] = lrow[r] * alpha + rs;
#pragma unroll
            for (int f = 0; f < 8; ++f) oacc[f][r] *= alpha;
        }

        // ---- P -> per-wave LDS tile (re-layout to A fragments) ----
#pragma unroll
        for (int f = 0; f < 8; ++f)
#pragma unroll
            for (int r = 0; r < 8; ++r)
                Pw[(r + hi * 8) * 136 + f * 16 + lo16] = (__bf16)sc[f][r];

        // ---- O += P * V ----
#pragma unroll
        for (int ks = 0; ks < 4; ++ks) {
            FragAB pa;
            const __bf16* pp = Pw + lo16 * 136 + ks * 32 + hi * 8;
            pa.u[0] = *(const uint4*)(pp);
            pa.u[1] = *(const uint4*)(pp + 16);
#pragma unroll
            for (int f2 = 0; f2 < 8; ++f2) {
                FragAB vb;
                const __bf16* vp = Vt + (f2 * 16 + lo16) * 136 + ks * 32 + hi * 8;
                vb.u[0] = *(const uint4*)(vp);
                vb.u[1] = *(const uint4*)(vp + 16);
                oacc[f2] = wmma_bf16(pa, vb, oacc[f2]);
            }
        }
    }

    // ---- normalize + store context [b, s, h*128+d] ----
#pragma unroll
    for (int f2 = 0; f2 < 8; ++f2) {
#pragma unroll
        for (int r = 0; r < 8; ++r) {
            const int qrow = q0 + wave * 16 + r + hi * 8;
            const int d    = f2 * 16 + lo16;
            const float o  = oacc[f2][r] / lrow[r];
            ctx[(size_t)(b * SEQ + qrow) * D_MODEL + h * HEAD_DIM + d] = (__bf16)o;
        }
    }
}

// ---------------------------------------------------------------------------
extern "C" void kernel_launch(void* const* d_in, const int* in_sizes, int n_in,
                              void* d_out, int out_size, void* d_ws, size_t ws_size,
                              hipStream_t stream)
{
    (void)in_sizes; (void)n_in; (void)out_size; (void)ws_size;

    const float* q  = (const float*)d_in[0];
    const float* k  = (const float*)d_in[1];
    const float* v  = (const float*)d_in[2];
    const float* Wq = (const float*)d_in[3];
    const float* bq = (const float*)d_in[4];
    const float* Wk = (const float*)d_in[5];
    const float* bk = (const float*)d_in[6];
    const float* Wv = (const float*)d_in[7];
    const float* bv = (const float*)d_in[8];
    const float* Wo = (const float*)d_in[9];
    const float* bo = (const float*)d_in[10];
    float* out = (float*)d_out;

    const int M = BATCH * SEQ;                   // 4096
    const size_t MD = (size_t)M * D_MODEL;       // 8 Mi elems
    const size_t DD = (size_t)D_MODEL * D_MODEL; // 4 Mi
    const size_t HD = (size_t)HEAD_DIM * D_MODEL;
    const size_t MH = (size_t)M * HEAD_DIM;

    // workspace (all bf16)
    __bf16* qb  = (__bf16*)d_ws;
    __bf16* kb  = qb  + MD;
    __bf16* vb  = kb  + MD;
    __bf16* WqT = vb  + MD;
    __bf16* WkT = WqT + DD;
    __bf16* WvT = WkT + HD;
    __bf16* WoT = WvT + HD;
    __bf16* qh  = WoT + DD;
    __bf16* kh  = qh  + MD;
    __bf16* vhT = kh  + MH;   // [B][HEAD_DIM][SEQ]
    __bf16* ctx = vhT + MH;

    dim3 blk(256);
    dim3 tblk(32, 8);

    // one-time conversions
    const int n4 = (int)(MD / 4);
    cvt_f32_to_bf16<<<n4 / 256, blk, 0, stream>>>(q, qb, n4);
    cvt_f32_to_bf16<<<n4 / 256, blk, 0, stream>>>(k, kb, n4);
    cvt_f32_to_bf16<<<n4 / 256, blk, 0, stream>>>(v, vb, n4);
    transpose_cvt_bf16<<<dim3(D_MODEL / 32, D_MODEL / 32), tblk, 0, stream>>>(Wq, WqT, D_MODEL, D_MODEL);
    transpose_cvt_bf16<<<dim3(HEAD_DIM / 32, D_MODEL / 32), tblk, 0, stream>>>(Wk, WkT, D_MODEL, HEAD_DIM);
    transpose_cvt_bf16<<<dim3(HEAD_DIM / 32, D_MODEL / 32), tblk, 0, stream>>>(Wv, WvT, D_MODEL, HEAD_DIM);
    transpose_cvt_bf16<<<dim3(D_MODEL / 32, D_MODEL / 32), tblk, 0, stream>>>(Wo, WoT, D_MODEL, D_MODEL);

    // projections
    gemm_bf16_wmma<__bf16, false>
        <<<dim3(D_MODEL / 128, M / 128), blk, 0, stream>>>(qb, WqT, bq, qh, M, D_MODEL, D_MODEL);
    gemm_bf16_wmma<__bf16, false>
        <<<dim3(HEAD_DIM / 128, M / 128), blk, 0, stream>>>(kb, WkT, bk, kh, M, HEAD_DIM, D_MODEL);
    gemm_bf16_wmma<__bf16, true>
        <<<dim3(HEAD_DIM / 128, M / 128), blk, 0, stream>>>(vb, WvT, bv, vhT, M, HEAD_DIM, D_MODEL);

    // flash MQA attention
    const size_t smem_bytes = (size_t)(2 * 128 * 136 + 8 * 16 * 136) * sizeof(__bf16);
    mqa_attention_wmma
        <<<dim3(SEQ / 128, NUM_HEADS, BATCH), blk, smem_bytes, stream>>>(qh, kh, vhT, ctx);

    // output projection
    gemm_bf16_wmma<float, false>
        <<<dim3(D_MODEL / 128, M / 128), blk, 0, stream>>>(ctx, WoT, bo, out, M, D_MODEL, D_MODEL);
}
